// SparsePromptFull_67697274520027
// MI455X (gfx1250) — matrix operand order repaired and checked by reading, not compiled, not run on hardware
//
#include <hip/hip_runtime.h>
#include <hip/hip_bf16.h>

// ---------------------------------------------------------------------------
// SparsePromptFull on MI455X (gfx1250, wave32).
//
// Roofline: all tensors are L2-resident (P_lr 2.4MB, edge/occ 1.2MB each,
// 192MB L2, 23.3TB/s HBM) -> latency bound, not bandwidth bound. The heavy
// stage is the trilinear upsample + D-reductions (295K pixels x 128 depths).
// The D-axis resize is a fixed 128x32 linear map whose contraction dim K=32
// matches one v_wmma_f32_16x16x32_f16 exactly, so one wave computes a
// 128x16 (D x pixels) block with 8 WMMAs, then reduces in registers.
// The 8-tile loop is fully unrolled so WMMA (TRANS-tracked on CDNA5)
// co-executes with the neighboring tiles' A-matrix VALU construction.
// Diffusion runs as 3 launches of 10 fused Jacobi iterations in 62KB LDS
// tiles (halo 10) instead of 30 global round trips.
// ---------------------------------------------------------------------------

typedef __attribute__((ext_vector_type(16))) _Float16 v16h;
typedef __attribute__((ext_vector_type(8)))  float    v8f;

#define HH   384
#define WW   768
#define HW_  (HH*WW)          // 294912
#define H4   96
#define W4   192
#define DLR  32
#define DHR  128
#define NBLK (HW_/256)        // 1152

#define TI   24               // Jacobi tile interior
#define HALO 10               // = iterations per launch
#define RG   (TI + 2*HALO)    // 44
#define RG2  (RG*RG)

// ---------------------------------------------------------------------------
// Stage A: fused trilinear upsample (WMMA on D axis) + per-pixel reductions.
// Grid (48,96), block 128 (4 waves). Wave w handles row y0+w, 16 pixels.
// ---------------------------------------------------------------------------
__global__ __launch_bounds__(128) void spf_prob(
    const float* __restrict__ P, const float* __restrict__ edge,
    const float* __restrict__ occ,
    float* __restrict__ conf, float* __restrict__ dhat, float* __restrict__ dinit)
{
    __shared__ float sP[DLR][3][6];    // LR patch: 3 rows x 6 cols x 32 depths
    const int x0 = blockIdx.x * 16;
    const int y0 = blockIdx.y * 4;
    const int tid = threadIdx.x;

    const int rl0 = (int)floorf((float)y0 * (95.0f/383.0f));
    const int cl0 = (int)floorf((float)x0 * (191.0f/767.0f));

    for (int i = tid; i < DLR*3*6; i += 128) {
        int k = i / 18, rm = i % 18, r = rm / 6, c = rm % 6;
        int gr = min(rl0 + r, H4-1), gc = min(cl0 + c, W4-1);
        sP[k][r][c] = P[(k*H4 + gr)*W4 + gc];
    }
    __syncthreads();

    const int wv = tid >> 5, lane = tid & 31;
    const int y = y0 + wv, pix = lane & 15, x = x0 + pix;

    __builtin_prefetch(&edge[y*WW + x], 0, 0);   // global_prefetch_b8
    __builtin_prefetch(&occ [y*WW + x], 0, 0);

    // align_corners bilinear weights (H,W) for this pixel
    float ph = (float)y * (95.0f/383.0f);
    int rl = min(max((int)floorf(ph), 0), H4-1);
    float wh = ph - (float)rl;
    int rh = min(rl + 1, H4-1);
    float pw = (float)x * (191.0f/767.0f);
    int cl = min(max((int)floorf(pw), 0), W4-1);
    float wwp = pw - (float)cl;
    int ch = min(cl + 1, W4-1);
    int r0 = rl - rl0, r1 = rh - rl0, c0 = cl - cl0, c1 = ch - cl0;
    float w00 = (1.f-wh)*(1.f-wwp), w01 = (1.f-wh)*wwp;
    float w10 = wh*(1.f-wwp),       w11 = wh*wwp;

    // B matrix (32x16 f16, K x N): lane n holds pixel n, K 0..15;
    // lane n+16 holds pixel n, K 16..31. Element e -> K = kb + e.
    const int kb = (lane >> 4) * 16;
    v16h B;
#pragma unroll
    for (int e = 0; e < 16; e++) {
        int k = kb + e;
        float v = w00*sP[k][r0][c0] + w01*sP[k][r0][c1]
                + w10*sP[k][r1][c0] + w11*sP[k][r1][c1];
        B[e] = (_Float16)v;
    }

    float S = 0.f, Sd = 0.f, m1 = -3.4e38f, m2 = -3.4e38f;
    int   a1 = 0;
    const int hiHalf = (lane >= 16) ? 1 : 0;

    // Fully unrolled: 8 static v_wmma so A-build / accumulator-drain VALU of
    // neighboring tiles co-executes in each WMMA's shadow (XDL co-exec).
#pragma unroll
    for (int t = 0; t < 8; t++) {
        // A matrix (16x32 f16): lanes hold row M = lane&15; global D row:
        int doA = t*16 + (lane & 15);
        float pd = (float)doA * (31.0f/127.0f);
        int lo = min(max((int)floorf(pd), 0), DLR-1);
        int hi = min(lo + 1, DLR-1);
        float wd = pd - (float)lo;
        v16h A;
#pragma unroll
        for (int e = 0; e < 16; e++) {
            // A layout matching B: lanes<16 K=0..15, lanes>=16 K=16..31
            int di = hiHalf * 16 + e;
            float a = 0.f;
            a += (di == lo) ? (1.0f - wd) : 0.f;
            a += (di == hi) ? wd          : 0.f;
            A[e] = (_Float16)a;
        }
        v8f c = {};
        c = __builtin_amdgcn_wmma_f32_16x16x32_f16(
                false, A, false, B, (short)0, c, false, false);

        // C/D layout: lane l<16 -> (M=r, N=l); l>=16 -> (M=r+8, N=l-16)
#pragma unroll
        for (int r = 0; r < 8; r++) {
            int   dc = t*16 + r + (hiHalf ? 8 : 0);
            float v  = c[r];
            S  += v;
            Sd += v * (float)dc;
            if (v > m1) { m2 = m1; m1 = v; a1 = dc; }
            else        { m2 = fmaxf(m2, v); }
        }
    }

    // merge lane pair (l, l+16) which together cover all 128 depths of pixel l&15
    float pS  = __shfl_xor(S, 16, 32);
    float pSd = __shfl_xor(Sd, 16, 32);
    float pm1 = __shfl_xor(m1, 16, 32);
    float pm2 = __shfl_xor(m2, 16, 32);
    int   pa1 = __shfl_xor(a1, 16, 32);
    S += pS; Sd += pSd;
    if (pm1 > m1)       { m2 = fmaxf(m1, pm2); m1 = pm1; a1 = pa1; }
    else if (pm1 == m1) { m2 = m1; a1 = min(a1, pa1); }
    else                { m2 = fmaxf(m2, pm1); }

    if (lane < 16) {
        float Sn   = S + 1e-6f;
        float pmax = m1 / Sn;
        float p2   = m2 / Sn;
        float psr  = pmax / (p2 + 1e-6f);
        int   idx  = y*WW + x;
        float ev = edge[idx], ov = occ[idx];
        float cf = pmax * tanhf(psr) * __expf(-2.0f * ev) * fmaxf(1.0f - ov, 0.0f);
        conf [idx] = cf;
        dhat [idx] = (float)a1;       // already in [0, DHR-1]
        dinit[idx] = Sd / Sn;
    }
}

// ---------------------------------------------------------------------------
// Stage B: NMS + deterministic top-k via histogram + index-ordered ranks.
// ---------------------------------------------------------------------------
__global__ void spf_zero(unsigned* hist, int* sel) {
    int i = blockIdx.x*256 + threadIdx.x;
    if (i < 1024) hist[i] = 0u;
    if (i < 4)    sel[i]  = 0;
}

__global__ void spf_nms(const float* __restrict__ conf,
                        unsigned* __restrict__ keep0, unsigned* __restrict__ hist) {
    int i = blockIdx.x*256 + threadIdx.x;
    int y = i / WW, x = i % WW;
    float c = conf[i];
    float p = -3.4e38f;
#pragma unroll
    for (int dy = -1; dy <= 1; dy++)
#pragma unroll
        for (int dx = -1; dx <= 1; dx++) {
            int yy = y + dy, xx = x + dx;
            if (yy >= 0 && yy < HH && xx >= 0 && xx < WW)
                p = fmaxf(p, conf[yy*WW + xx]);
        }
    unsigned k = (c >= 0.1f && c >= p) ? 1u : 0u;
    keep0[i] = k;
    if (k) {
        int b = min(1023, (int)(c * 1024.0f));
        atomicAdd(&hist[b], 1u);     // order-independent -> deterministic
    }
}

__global__ void spf_select(const unsigned* hist, const int* topk_p, int* sel) {
    if (blockIdx.x == 0 && threadIdx.x == 0) {
        int topk = topk_p[0];
        unsigned acc = 0; int bcut = -1, rem = 0, accept = 1;
        for (int b = 1023; b >= 0; b--) {
            unsigned nn = acc + hist[b];
            if ((int)nn > topk) { bcut = b; rem = topk - (int)acc; accept = 0; break; }
            acc = nn;
        }
        sel[0] = bcut; sel[1] = rem; sel[2] = accept;
    }
}

__global__ void spf_bcnt(const float* conf, const unsigned* keep0,
                         const int* sel, unsigned* blkCnt) {
    __shared__ unsigned wsum[8];
    int i = blockIdx.x*256 + threadIdx.x;
    int bcut = sel[0], accept = sel[2];
    float c = conf[i];
    int bin = min(1023, (int)(c * 1024.0f));
    int pred = (!accept) && keep0[i] && (bin == bcut);
    unsigned long long m = __ballot(pred);
    int lane = threadIdx.x & 31, wv = threadIdx.x >> 5;
    if (lane == 0) wsum[wv] = (unsigned)__popcll(m);
    __syncthreads();
    if (threadIdx.x == 0) {
        unsigned s = 0;
        for (int w = 0; w < 8; w++) s += wsum[w];
        blkCnt[blockIdx.x] = s;
    }
}

__global__ void spf_scan(const unsigned* blkCnt, unsigned* blkOff) {
    if (blockIdx.x == 0 && threadIdx.x == 0) {
        unsigned acc = 0;
        for (int b = 0; b < NBLK; b++) { blkOff[b] = acc; acc += blkCnt[b]; }
    }
}

__global__ void spf_fin(const float* conf, const unsigned* keep0, const int* sel,
                        const unsigned* blkOff, unsigned* keepF) {
    __shared__ unsigned woff[8];
    int i = blockIdx.x*256 + threadIdx.x;
    int bcut = sel[0], rem = sel[1], accept = sel[2];
    float c = conf[i];
    int bin = min(1023, (int)(c * 1024.0f));
    unsigned k0 = keep0[i];
    int pred = (!accept) && k0 && (bin == bcut);
    unsigned long long m = __ballot(pred);
    int lane = threadIdx.x & 31, wv = threadIdx.x >> 5;
    if (lane == 0) woff[wv] = (unsigned)__popcll(m);
    __syncthreads();
    if (threadIdx.x == 0) {
        unsigned a = 0;
        for (int w = 0; w < 8; w++) { unsigned t = woff[w]; woff[w] = a; a += t; }
    }
    __syncthreads();
    unsigned lanepre = (unsigned)__popcll(m & ((1ull << lane) - 1ull));
    unsigned rank = blkOff[blockIdx.x] + woff[wv] + lanepre;
    unsigned kf;
    if (accept) kf = k0;
    else kf = (k0 && (bin > bcut || (bin == bcut && (int)rank < rem))) ? 1u : 0u;
    keepF[i] = kf;
}

// ---------------------------------------------------------------------------
// Stage C: diffusion weights + fused Jacobi (10 iters/launch, halo-10 LDS tile)
// ---------------------------------------------------------------------------
__global__ void spf_weights(const float* edge, const float* occ, const float* conf,
                            const float* dhat, const float* dinit, const unsigned* keepF,
                            float* wu, float* wd, float* wl, float* wr,
                            float* num0, float* dinv, float* Ra) {
    int i = blockIdx.x*256 + threadIdx.x;
    int y = i / WW, x = i % WW;
    float e = edge[i], o = occ[i];
    int yu = max(y-1, 0), yd = min(y+1, HH-1);
    int xl = max(x-1, 0), xr = min(x+1, WW-1);
    float eu = edge[yu*WW + x], ou = occ[yu*WW + x];
    float ed = edge[yd*WW + x], od = occ[yd*WW + x];
    float el = edge[y*WW + xl], ol = occ[y*WW + xl];
    float er = edge[y*WW + xr], orr = occ[y*WW + xr];
    // LAMBDA_E=2: exp(-2*(e+en)/2) = exp(-(e+en))
    float u = __expf(-(e+eu)) * (1.f-o) * (1.f-ou);
    float d = __expf(-(e+ed)) * (1.f-o) * (1.f-od);
    float l = __expf(-(e+el)) * (1.f-o) * (1.f-ol);
    float r = __expf(-(e+er)) * (1.f-o) * (1.f-orr);
    float wsum = u + d + l + r + 1e-6f;
    float m = keepF[i] ? fminf(fmaxf(conf[i], 0.f), 1.f) : 0.f;   // valid==keepF here
    wu[i] = u; wd[i] = d; wl[i] = l; wr[i] = r;
    num0[i] = m * (dhat[i] - dinit[i]);
    dinv[i] = 1.0f / (m + 0.8f * wsum + 1e-6f);
    Ra[i]   = 0.f;
}

__global__ __launch_bounds__(256) void spf_jac(
    const float* wu, const float* wd, const float* wl, const float* wr,
    const float* num0, const float* dinv, const float* Rin, float* Rout) {
    __shared__ float s_wu[RG2], s_wd[RG2], s_wl[RG2], s_wr[RG2];
    __shared__ float s_n0[RG2], s_di[RG2], s_Ra[RG2], s_Rb[RG2];
    const int y0 = (int)blockIdx.y * TI - HALO;
    const int x0 = (int)blockIdx.x * TI - HALO;
    // clamp bounds in GLOBAL coords -> exact replicate at true image borders
    const int rlo = max(0, -y0),  rhi = min(RG-1, (HH-1) - y0);
    const int clo = max(0, -x0),  chi = min(RG-1, (WW-1) - x0);

    for (int i = threadIdx.x; i < RG2; i += 256) {
        int r = i / RG, c = i % RG;
        int gy = min(max(y0 + r, 0), HH-1);
        int gx = min(max(x0 + c, 0), WW-1);
        int g = gy*WW + gx;
        s_wu[i] = wu[g]; s_wd[i] = wd[g]; s_wl[i] = wl[g]; s_wr[i] = wr[g];
        s_n0[i] = num0[g]; s_di[i] = dinv[g]; s_Ra[i] = Rin[g];
    }
    __syncthreads();

    float* a = s_Ra; float* b = s_Rb;
    for (int it = 0; it < HALO; it++) {
        for (int i = threadIdx.x; i < RG2; i += 256) {
            int r = i / RG, c = i % RG;
            int ru = max(r-1, rlo), rd = min(r+1, rhi);
            int cl = max(c-1, clo), cr = min(c+1, chi);
            float nbr = s_wu[i]*a[ru*RG + c] + s_wd[i]*a[rd*RG + c]
                      + s_wl[i]*a[r*RG + cl] + s_wr[i]*a[r*RG + cr];
            b[i] = (s_n0[i] + 0.8f * nbr) * s_di[i];
        }
        __syncthreads();
        float* t = a; a = b; b = t;
    }
    for (int i = threadIdx.x; i < RG2; i += 256) {
        int r = i / RG, c = i % RG;
        if (r >= HALO && r < HALO+TI && c >= HALO && c < HALO+TI)
            Rout[(y0 + r)*WW + (x0 + c)] = a[i];
    }
}

__global__ void spf_out(const float* dinit, const float* R, float* out) {
    int i = blockIdx.x*256 + threadIdx.x;
    out[i] = fmaxf(dinit[i] + R[i], 0.f);
}

// ---------------------------------------------------------------------------
extern "C" void kernel_launch(void* const* d_in, const int* in_sizes, int n_in,
                              void* d_out, int out_size, void* d_ws, size_t ws_size,
                              hipStream_t stream) {
    const float* P    = (const float*)d_in[0];   // [1,32,96,192]
    const float* edge = (const float*)d_in[1];   // [1,1,384,768]
    const float* occ  = (const float*)d_in[2];   // [1,1,384,768]
    const int*   topk = (const int*)  d_in[5];   // scalar, read device-side
    float* out = (float*)d_out;                  // [1,1,384,768]

    // workspace layout (floats); ~15.4 MB total
    float* ws = (float*)d_ws;
    float *conf = ws,            *dhat = ws +  1*(size_t)HW_, *dinit = ws + 2*(size_t)HW_;
    float *wu = ws + 3*(size_t)HW_, *wdn = ws + 4*(size_t)HW_;
    float *wlf = ws + 5*(size_t)HW_, *wrt = ws + 6*(size_t)HW_;
    float *n0 = ws + 7*(size_t)HW_, *di = ws + 8*(size_t)HW_;
    float *Ra = ws + 9*(size_t)HW_, *Rb = ws + 10*(size_t)HW_;
    unsigned* keep0  = (unsigned*)(ws + 11*(size_t)HW_);
    unsigned* keepF  = (unsigned*)(ws + 12*(size_t)HW_);
    unsigned* hist   = (unsigned*)(ws + 13*(size_t)HW_);
    int*      sel    = (int*)     (ws + 13*(size_t)HW_ + 1024);
    unsigned* blkCnt = (unsigned*)(ws + 13*(size_t)HW_ + 1032);
    unsigned* blkOff = blkCnt + NBLK;

    spf_zero   <<<4, 256, 0, stream>>>(hist, sel);
    spf_prob   <<<dim3(WW/16, HH/4), 128, 0, stream>>>(P, edge, occ, conf, dhat, dinit);
    spf_nms    <<<NBLK, 256, 0, stream>>>(conf, keep0, hist);
    spf_select <<<1, 32, 0, stream>>>(hist, topk, sel);
    spf_bcnt   <<<NBLK, 256, 0, stream>>>(conf, keep0, sel, blkCnt);
    spf_scan   <<<1, 32, 0, stream>>>(blkCnt, blkOff);
    spf_fin    <<<NBLK, 256, 0, stream>>>(conf, keep0, sel, blkOff, keepF);
    spf_weights<<<NBLK, 256, 0, stream>>>(edge, occ, conf, dhat, dinit, keepF,
                                          wu, wdn, wlf, wrt, n0, di, Ra);
    // 30 Jacobi iterations = 3 x (10 fused iterations in LDS)
    spf_jac    <<<dim3(WW/TI, HH/TI), 256, 0, stream>>>(wu, wdn, wlf, wrt, n0, di, Ra, Rb);
    spf_jac    <<<dim3(WW/TI, HH/TI), 256, 0, stream>>>(wu, wdn, wlf, wrt, n0, di, Rb, Ra);
    spf_jac    <<<dim3(WW/TI, HH/TI), 256, 0, stream>>>(wu, wdn, wlf, wrt, n0, di, Ra, Rb);
    spf_out    <<<NBLK, 256, 0, stream>>>(dinit, Rb, out);

    (void)in_sizes; (void)n_in; (void)out_size; (void)ws_size;
}
